// Transform_Network_83700322665231
// MI455X (gfx1250) — matrix operand order repaired and checked by reading, not compiled, not run on hardware
//
#include <hip/hip_runtime.h>
#include <hip/hip_bf16.h>
#include <stdint.h>

// ---------------------------------------------------------------------------
// Fused MoE-style MLP for MI455X (gfx1250, wave32, WMMA bf16 16x16x32)
//   latent_pi = select(flag==0, relu(x@Wt1+bt1)@Wt2+bt2,
//                                relu(x@Wa1+ba1)@Wa2+ba2)
//   latent_vf = relu(relu(x@Wv1+bv1)@Wv2+bv2)
// Each wave owns M=32 rows (2 WMMA M-tiles) so every B fragment feeds 2 WMMAs,
// halving L2 weight traffic vs the M=16 version.
// ---------------------------------------------------------------------------

typedef __bf16 v16bf __attribute__((ext_vector_type(16)));
typedef __bf16 v4bf  __attribute__((ext_vector_type(4)));
typedef float  v8f   __attribute__((ext_vector_type(8)));

#define N_ROWS 262144
#define D_IN   512
#define H_DIM  128
#define THREADS 128
#define WAVES   4
#define ROWS_PER_WAVE  32
#define ROWS_PER_BLOCK 128
#define SMEM_BYTES (ROWS_PER_BLOCK * D_IN * 2 + WAVES * ROWS_PER_WAVE * H_DIM * 2) // 160 KB

union AFrag { uint4 u[2]; v16bf v; };

// A-matrix fragment from LDS (16 rows x 32 K, row-major rows):
// lane (r16, sel): elements 0..15 = K off0..off0+15
// with off0 = k0 + sel*16
__device__ __forceinline__ v16bf load_frag_a(const __bf16* rowp, int off0) {
    AFrag f;
    f.u[0] = *(const uint4*)(rowp + off0);
    f.u[1] = *(const uint4*)(rowp + off0 + 8);
    return f.v;
}

// B-matrix fragment: column-major weights; lane holds 16 contiguous K values
// of its column starting at k0 + sel*16 (same contiguous k order as A)
__device__ __forceinline__ v16bf load_frag_b(const __bf16* p) {
    AFrag f;
    f.u[0] = *(const uint4*)(p);
    f.u[1] = *(const uint4*)(p + 8);
    return f.v;
}

__device__ __forceinline__ v8f wmma_bf16(v16bf a, v16bf b, v8f c) {
    return __builtin_amdgcn_wmma_f32_16x16x32_bf16(false, a, false, b,
                                                   (short)0, c, false, false);
}

// ---------------- prologue: fp32 row-major -> bf16 col-major weights --------
__global__ __launch_bounds__(256) void wconv_kernel(
    const float* __restrict__ Wt1, const float* __restrict__ Wa1,
    const float* __restrict__ Wv1, const float* __restrict__ Wt2,
    const float* __restrict__ Wa2, const float* __restrict__ Wv2,
    __bf16* __restrict__ W1T, __bf16* __restrict__ W2T)
{
    const int n1 = 3 * H_DIM * D_IN;      // 196608
    const int n2 = 3 * H_DIM * H_DIM;     // 49152
    int idx = blockIdx.x * 256 + threadIdx.x;
    if (idx < n1) {
        int ng = idx / D_IN, k = idx % D_IN;      // ng = e*128 + n
        int e = ng / H_DIM, n = ng % H_DIM;
        const float* W = (e == 0) ? Wt1 : (e == 1) ? Wa1 : Wv1; // [512][128]
        W1T[idx] = (__bf16)W[k * H_DIM + n];
    } else if (idx < n1 + n2) {
        int j = idx - n1;
        int ng = j / H_DIM, k = j % H_DIM;
        int e = ng / H_DIM, n = ng % H_DIM;
        const float* W = (e == 0) ? Wt2 : (e == 1) ? Wa2 : Wv2; // [128][128]
        W2T[j] = (__bf16)W[k * H_DIM + n];
    }
}

// ---------------- main fused kernel ----------------------------------------
__global__ __launch_bounds__(THREADS) void fused_moe_mlp(
    const float*  __restrict__ x,
    const __bf16* __restrict__ W1T,   // [3*128][512] n-major
    const __bf16* __restrict__ W2T,   // [3*128][128] n-major
    const float* __restrict__ bt1, const float* __restrict__ bt2,
    const float* __restrict__ ba1, const float* __restrict__ ba2,
    const float* __restrict__ bv1, const float* __restrict__ bv2,
    float* __restrict__ out)          // [pi: N*128][vf: N*128]
{
    extern __shared__ char smem[];
    __bf16* xs = (__bf16*)smem;                                 // [128][512]
    __bf16* hs = (__bf16*)(smem + ROWS_PER_BLOCK * D_IN * 2);   // [4][32][128]

    const int tid  = threadIdx.x;
    const int wave = tid >> 5;
    const int lane = tid & 31;
    const int r16  = lane & 15;
    const int sel  = lane >> 4;

    const int rowBlock = blockIdx.x * ROWS_PER_BLOCK;
    const int rowWave  = wave * ROWS_PER_WAVE;

    // stage this wave's 32 rows: fp32 global -> bf16 LDS (coalesced b128)
    for (int r = 0; r < ROWS_PER_WAVE; ++r) {
        const float4* src = (const float4*)(x + (size_t)(rowBlock + rowWave + r) * D_IN);
        __bf16* dst = xs + (rowWave + r) * D_IN;
        #pragma unroll
        for (int c = 0; c < 4; ++c) {
            int i = lane + c * 32;              // 0..127 float4 chunks
            float4 v = src[i];
            v4bf p = { (__bf16)v.x, (__bf16)v.y, (__bf16)v.z, (__bf16)v.w };
            *(v4bf*)(dst + i * 4) = p;
        }
    }
    __syncthreads();

    // routing flags for the rows this lane owns in C-layout (M = g + 8*sel)
    float flg[2][8];
    #pragma unroll
    for (int m = 0; m < 2; ++m)
        #pragma unroll
        for (int g = 0; g < 8; ++g)
            flg[m][g] = (float)xs[(rowWave + m * 16 + g + 8 * sel) * D_IN + (D_IN - 1)];

    const __bf16* arow0 = xs + (rowWave + r16) * D_IN;        // M-tile 0
    const __bf16* arow1 = xs + (rowWave + 16 + r16) * D_IN;   // M-tile 1
    __bf16*       hw    = hs + wave * (ROWS_PER_WAVE * H_DIM);
    const __bf16* hrow0 = hw + r16 * H_DIM;
    const __bf16* hrow1 = hw + (16 + r16) * H_DIM;

    float* out_pi = out;
    float* out_vf = out + (size_t)N_ROWS * H_DIM;

    #pragma unroll 1
    for (int e = 0; e < 3; ++e) {
        const float*  b1 = (e == 0) ? bt1 : (e == 1) ? ba1 : bv1;
        const float*  b2 = (e == 0) ? bt2 : (e == 1) ? ba2 : bv2;
        const __bf16* w1 = W1T + (size_t)e * H_DIM * D_IN;
        const __bf16* w2 = W2T + (size_t)e * H_DIM * H_DIM;

        // ------- layer 1: h = relu(x @ W1 + b1)   [32x512]@[512x128] -------
        #pragma unroll 1
        for (int ng = 0; ng < 2; ++ng) {             // 2 groups of 4 N-tiles
            v8f acc[2][4];
            #pragma unroll
            for (int j = 0; j < 4; ++j) {
                float bv = b1[(ng * 4 + j) * 16 + r16];
                #pragma unroll
                for (int i = 0; i < 8; ++i) { acc[0][j][i] = bv; acc[1][j][i] = bv; }
            }
            for (int k0 = 0; k0 < D_IN; k0 += 32) {
                v16bf a0 = load_frag_a(arow0, k0 + sel * 16);
                v16bf a1 = load_frag_a(arow1, k0 + sel * 16);
                #pragma unroll
                for (int j = 0; j < 4; ++j) {
                    v16bf b = load_frag_b(w1 + (size_t)((ng * 4 + j) * 16 + r16) * D_IN
                                             + k0 + sel * 16);
                    acc[0][j] = wmma_bf16(a0, b, acc[0][j]);   // B reused for
                    acc[1][j] = wmma_bf16(a1, b, acc[1][j]);   // both M-tiles
                }
            }
            // relu -> bf16 -> per-wave LDS scratch (C layout -> row-major h)
            #pragma unroll
            for (int j = 0; j < 4; ++j)
                #pragma unroll
                for (int m = 0; m < 2; ++m)
                    #pragma unroll
                    for (int g = 0; g < 8; ++g)
                        hw[(m * 16 + g + 8 * sel) * H_DIM + (ng * 4 + j) * 16 + r16] =
                            (__bf16)fmaxf(acc[m][j][g], 0.0f);
        }
        __syncthreads();   // fence LDS store->load across type-punned access

        // ------- layer 2: y = h @ W2 + b2   [32x128]@[128x128] -------------
        #pragma unroll 1
        for (int ng = 0; ng < 2; ++ng) {
            v8f acc2[2][4];
            #pragma unroll
            for (int j = 0; j < 4; ++j) {
                float bv = b2[(ng * 4 + j) * 16 + r16];
                #pragma unroll
                for (int i = 0; i < 8; ++i) { acc2[0][j][i] = bv; acc2[1][j][i] = bv; }
            }
            for (int k0 = 0; k0 < H_DIM; k0 += 32) {
                v16bf a0 = load_frag_a(hrow0, k0 + sel * 16);
                v16bf a1 = load_frag_a(hrow1, k0 + sel * 16);
                #pragma unroll
                for (int j = 0; j < 4; ++j) {
                    v16bf b = load_frag_b(w2 + (size_t)((ng * 4 + j) * 16 + r16) * H_DIM
                                             + k0 + sel * 16);
                    acc2[0][j] = wmma_bf16(a0, b, acc2[0][j]);
                    acc2[1][j] = wmma_bf16(a1, b, acc2[1][j]);
                }
            }
            // ------- epilogue ----------------------------------------------
            #pragma unroll
            for (int j = 0; j < 4; ++j) {
                #pragma unroll
                for (int m = 0; m < 2; ++m) {
                    #pragma unroll
                    for (int g = 0; g < 8; ++g) {
                        size_t row = (size_t)(rowBlock + rowWave + m * 16 + g + 8 * sel);
                        size_t o   = row * H_DIM + (ng * 4 + j) * 16 + r16;
                        float  v   = acc2[m][j][g];
                        if (e == 0) {
                            out_pi[o] = v;                            // transform
                        } else if (e == 1) {
                            if (flg[m][g] != 0.0f) out_pi[o] = v;     // actor
                        } else {
                            out_vf[o] = fmaxf(v, 0.0f);               // critic
                        }
                    }
                }
            }
        }
        __syncthreads();   // protect per-wave h scratch reuse next iteration
    }
}

// ---------------------------------------------------------------------------
extern "C" void kernel_launch(void* const* d_in, const int* in_sizes, int n_in,
                              void* d_out, int out_size, void* d_ws, size_t ws_size,
                              hipStream_t stream)
{
    const float* x   = (const float*)d_in[0];
    const float* Wt1 = (const float*)d_in[1];
    const float* bt1 = (const float*)d_in[2];
    const float* Wt2 = (const float*)d_in[3];
    const float* bt2 = (const float*)d_in[4];
    const float* Wa1 = (const float*)d_in[5];
    const float* ba1 = (const float*)d_in[6];
    const float* Wa2 = (const float*)d_in[7];
    const float* ba2 = (const float*)d_in[8];
    const float* Wv1 = (const float*)d_in[9];
    const float* bv1 = (const float*)d_in[10];
    const float* Wv2 = (const float*)d_in[11];
    const float* bv2 = (const float*)d_in[12];

    __bf16* W1T = (__bf16*)d_ws;                    // 3*128*512 bf16
    __bf16* W2T = W1T + 3 * H_DIM * D_IN;           // 3*128*128 bf16

    const int wtotal = 3 * H_DIM * D_IN + 3 * H_DIM * H_DIM;   // 245760
    wconv_kernel<<<(wtotal + 255) / 256, 256, 0, stream>>>(
        Wt1, Wa1, Wv1, Wt2, Wa2, Wv2, W1T, W2T);

    (void)hipFuncSetAttribute((const void*)fused_moe_mlp,
                              hipFuncAttributeMaxDynamicSharedMemorySize,
                              (int)SMEM_BYTES);

    fused_moe_mlp<<<N_ROWS / ROWS_PER_BLOCK, THREADS, SMEM_BYTES, stream>>>(
        x, W1T, W2T, bt1, bt2, ba1, ba2, bv1, bv2, (float*)d_out);
}
